// AnimationPredictor_24696061952410
// MI455X (gfx1250) — matrix operand order repaired and checked by reading, not compiled, not run on hardware
//
#include <hip/hip_runtime.h>
#include <math.h>

typedef __attribute__((ext_vector_type(16))) _Float16 v16h;
typedef __attribute__((ext_vector_type(8)))  float    v8f;
typedef unsigned int u32;
typedef __attribute__((ext_vector_type(4))) u32   u32x4;
typedef __attribute__((ext_vector_type(4))) float f32x4;

#define BATCH_N  524288
#define IN_F     128
#define H1_F     256
#define O1_F     10
#define O2_F     6
#define W2IN     138          // IN + O1

#define WAVES    8
#define NTHREADS (WAVES * 32)
#define NTILES   (BATCH_N / 16)
#define NPAIRS   (NTILES / 2)     // each wave handles 2 row-tiles (M=32)
#define NBLOCKS  1024

// ---- dynamic LDS layout (bytes) ----
#define OFF_W1P   0           // 16*4*32*8 u32  = 65536 B  (W1  B-frags, f16 pairs)
#define OFF_W2P   65536       // 65536 B                   (W2x B-frags)
#define OFF_WO1P  131072      // 8*32*8 u32     = 8192 B   (Wo1 padded 10->16)
#define OFF_WO2P  139264      // 8192 B                    (Wo2 padded 6->16)
#define OFF_W2Y   147456      // 10*256 f32     = 10240 B  (one-hot columns of W2, f32)
#define OFF_B1    157696      // 1024 B
#define OFF_B2    158720      // 1024 B
#define OFF_BO1   159744      // 64 B (padded to 16)
#define OFF_BO2   159808      // 64 B
#define OFF_H     159872      // per-wave h staging, 32 rows x HSTRIDE halves
#define HSTRIDE   264         // 256 + 8 halves pad: 528 B/row = 132 dwords -> 4-bank rotation, conflict-free A loads
#define SMEM_BYTES (OFF_H + WAVES * 32 * HSTRIDE * 2)   // 159872 + 135168 = 295040

union ABFrag { u32 u[8]; u32x4 q[2]; v16h h; };

__device__ __forceinline__ u32 pkh(float x, float y) {
    unsigned short a = __builtin_bit_cast(unsigned short, (_Float16)x);
    unsigned short b = __builtin_bit_cast(unsigned short, (_Float16)y);
    return (u32)a | ((u32)b << 16);
}
__device__ __forceinline__ v8f bcast8(float x) {
    v8f c = {x, x, x, x, x, x, x, x};
    return c;
}
__device__ __forceinline__ float relu0(float x) { return x > 0.f ? x : 0.f; }
__device__ __forceinline__ v8f wmma_f16(v16h a, v16h b, v8f c) {
    return __builtin_amdgcn_wmma_f32_16x16x32_f16(false, a, false, b, (short)0, c, false, false);
}

__global__ __launch_bounds__(NTHREADS, 1)
void fused_mlp_kernel(const float* __restrict__ X,
                      const float* __restrict__ W1,  const float* __restrict__ b1,
                      const float* __restrict__ Wo1, const float* __restrict__ bo1,
                      const float* __restrict__ W2,  const float* __restrict__ b2,
                      const float* __restrict__ Wo2, const float* __restrict__ bo2,
                      float* __restrict__ out)
{
    extern __shared__ unsigned char smem[];
    u32*      sW1p  = (u32*)     (smem + OFF_W1P);
    u32*      sW2p  = (u32*)     (smem + OFF_W2P);
    u32*      sWo1p = (u32*)     (smem + OFF_WO1P);
    u32*      sWo2p = (u32*)     (smem + OFF_WO2P);
    float*    sW2y  = (float*)   (smem + OFF_W2Y);
    float*    sB1   = (float*)   (smem + OFF_B1);
    float*    sB2   = (float*)   (smem + OFF_B2);
    float*    sBo1  = (float*)   (smem + OFF_BO1);
    float*    sBo2  = (float*)   (smem + OFF_BO2);
    _Float16* sH    = (_Float16*)(smem + OFF_H);

    const int tid = threadIdx.x;

    // ---- pre-swizzle all weights into LDS (native WMMA B-layout, f16 pairs) ----
    // B 32x16 f16 layout: lanes 0-15 hold K=0..7,16..23, lanes 16-31 K=8..15,24..31 (VGPR v -> K=8h+16*(v>>2)+2*(v&3), +1).
    for (int p = tid; p < 16 * 4 * 32 * 8; p += NTHREADS) {
        int v = p & 7, lane = (p >> 3) & 31, ks = (p >> 8) & 3, nt = p >> 10;
        int n = nt * 16 + (lane & 15);
        int k = ks * 32 + (lane >> 4) * 8 + (v >> 2) * 16 + 2 * (v & 3);
        sW1p[p] = pkh(W1[n * IN_F + k], W1[n * IN_F + k + 1]);
        sW2p[p] = pkh(W2[n * W2IN + k], W2[n * W2IN + k + 1]);
    }
    for (int p = tid; p < 8 * 32 * 8; p += NTHREADS) {
        int v = p & 7, lane = (p >> 3) & 31, ks = p >> 8;
        int o = lane & 15;
        int k = ks * 32 + (lane >> 4) * 8 + (v >> 2) * 16 + 2 * (v & 3);
        sWo1p[p] = (o < O1_F) ? pkh(Wo1[o * H1_F + k], Wo1[o * H1_F + k + 1]) : 0u;
        sWo2p[p] = (o < O2_F) ? pkh(Wo2[o * H1_F + k], Wo2[o * H1_F + k + 1]) : 0u;
    }
    for (int p = tid; p < O1_F * 256; p += NTHREADS) {
        int o = p >> 8, n = p & 255;
        sW2y[o * 256 + n] = W2[n * W2IN + 128 + o];   // one-hot columns, exact f32
    }
    for (int p = tid; p < 256; p += NTHREADS) { sB1[p] = b1[p]; sB2[p] = b2[p]; }
    if (tid < 16) {
        sBo1[tid] = (tid < O1_F) ? bo1[tid] : 0.f;
        sBo2[tid] = (tid < O2_F) ? bo2[tid] : 0.f;
    }
    __syncthreads();

    const int wave = tid >> 5;
    const int lane = tid & 31;
    const int r  = lane & 15;      // row (A) / column (B,C,D) within 16
    const int kb = lane >> 4;      // half-select: A K-subblock, D row-half
    _Float16* myH = sH + wave * (32 * HSTRIDE);

    for (int tp = blockIdx.x * WAVES + wave; tp < NPAIRS; tp += gridDim.x * WAVES) {
        const float* Xt = X + (size_t)tp * (32 * IN_F);

        // ---- load 32-row X block once: f32 -> f16 in WMMA A-layout, reused by both GEMMs ----
        ABFrag a[2][4];
        #pragma unroll
        for (int t = 0; t < 2; t++) {
            #pragma unroll
            for (int ks = 0; ks < 4; ks++) {
                const float* xr = Xt + (t * 16 + r) * IN_F + ks * 32 + kb * 8;
                f32x4 x0 = *(const f32x4*)(xr);
                f32x4 x1 = *(const f32x4*)(xr + 4);
                f32x4 x2 = *(const f32x4*)(xr + 16);
                f32x4 x3 = *(const f32x4*)(xr + 20);
                a[t][ks].u[0] = pkh(x0.x, x0.y); a[t][ks].u[1] = pkh(x0.z, x0.w);
                a[t][ks].u[2] = pkh(x1.x, x1.y); a[t][ks].u[3] = pkh(x1.z, x1.w);
                a[t][ks].u[4] = pkh(x2.x, x2.y); a[t][ks].u[5] = pkh(x2.z, x2.w);
                a[t][ks].u[6] = pkh(x3.x, x3.y); a[t][ks].u[7] = pkh(x3.z, x3.w);
            }
        }

        // ---- stage 1: h1 = relu(X @ W1^T + b1) -> LDS (f16); B fragment shared by both tiles ----
        for (int nt = 0; nt < 16; nt++) {
            int n = nt * 16 + r;
            v8f c0 = bcast8(sB1[n]);
            v8f c1 = c0;
            #pragma unroll
            for (int ks = 0; ks < 4; ks++) {
                ABFrag b;
                const u32* bp = sW1p + ((nt * 4 + ks) * 32 + lane) * 8;
                b.q[0] = *(const u32x4*)(bp);
                b.q[1] = *(const u32x4*)(bp + 4);
                c0 = wmma_f16(a[0][ks].h, b.h, c0);
                c1 = wmma_f16(a[1][ks].h, b.h, c1);
            }
            #pragma unroll
            for (int v = 0; v < 8; v++) {          // D: lane holds col n, rows v+8*kb
                myH[(v + kb * 8) * HSTRIDE + n]      = (_Float16)relu0(c0[v]);
                myH[(16 + v + kb * 8) * HSTRIDE + n] = (_Float16)relu0(c1[v]);
            }
        }

        // ---- logits1 = h1 @ Wo1^T + bo1 (padded to 16 cols), per-row argmax -> one-hot ----
        int idxv[2][8];
        {
            v8f c0 = bcast8(sBo1[r]);
            v8f c1 = c0;
            #pragma unroll
            for (int ks = 0; ks < 8; ks++) {
                int k0 = ks * 32 + kb * 8;
                ABFrag a0, a1, b;
                const _Float16* hp0 = myH + r * HSTRIDE + k0;
                const _Float16* hp1 = myH + (16 + r) * HSTRIDE + k0;
                a0.q[0] = *(const u32x4*)(hp0);
                a0.q[1] = *(const u32x4*)(hp0 + 16);
                a1.q[0] = *(const u32x4*)(hp1);
                a1.q[1] = *(const u32x4*)(hp1 + 16);
                const u32* bp = sWo1p + (ks * 32 + lane) * 8;
                b.q[0] = *(const u32x4*)(bp);
                b.q[1] = *(const u32x4*)(bp + 4);
                c0 = wmma_f16(a0.h, b.h, c0);
                c1 = wmma_f16(a1.h, b.h, c1);
            }
            #pragma unroll
            for (int t = 0; t < 2; t++) {
                #pragma unroll
                for (int v = 0; v < 8; v++) {
                    float val = (r < O1_F) ? (t ? c1[v] : c0[v]) : -__builtin_inff();
                    int   idx = r;
                    #pragma unroll
                    for (int m = 1; m < 16; m <<= 1) {   // butterfly within 16-lane half
                        float ov = __shfl_xor(val, m, 32);
                        int   oi = __shfl_xor(idx, m, 32);
                        if (ov > val || (ov == val && oi < idx)) { val = ov; idx = oi; }
                    }
                    idxv[t][v] = idx;                    // identical across the 16-lane half
                    if (r < O1_F)
                        out[(size_t)(tp * 32 + t * 16 + v + kb * 8) * 16 + r] =
                            (idx == r) ? 1.f : 0.f;
                }
            }
        }

        // ---- stage 2: h2 = relu(X @ W2x^T + W2y[:,idx] + b2) -> LDS (f16) ----
        for (int nt = 0; nt < 16; nt++) {
            int n = nt * 16 + r;
            v8f c0 = bcast8(sB2[n]);
            v8f c1 = c0;
            #pragma unroll
            for (int ks = 0; ks < 4; ks++) {
                ABFrag b;
                const u32* bp = sW2p + ((nt * 4 + ks) * 32 + lane) * 8;
                b.q[0] = *(const u32x4*)(bp);
                b.q[1] = *(const u32x4*)(bp + 4);
                c0 = wmma_f16(a[0][ks].h, b.h, c0);
                c1 = wmma_f16(a[1][ks].h, b.h, c1);
            }
            #pragma unroll
            for (int v = 0; v < 8; v++) {            // one-hot column gather-add, then relu
                float h0 = c0[v] + sW2y[idxv[0][v] * 256 + n];
                float h1 = c1[v] + sW2y[idxv[1][v] * 256 + n];
                myH[(v + kb * 8) * HSTRIDE + n]      = (_Float16)relu0(h0);
                myH[(16 + v + kb * 8) * HSTRIDE + n] = (_Float16)relu0(h1);
            }
        }

        // ---- y2 = sigmoid(h2 @ Wo2^T + bo2) ----
        {
            v8f c0 = bcast8(sBo2[r]);
            v8f c1 = c0;
            #pragma unroll
            for (int ks = 0; ks < 8; ks++) {
                int k0 = ks * 32 + kb * 8;
                ABFrag a0, a1, b;
                const _Float16* hp0 = myH + r * HSTRIDE + k0;
                const _Float16* hp1 = myH + (16 + r) * HSTRIDE + k0;
                a0.q[0] = *(const u32x4*)(hp0);
                a0.q[1] = *(const u32x4*)(hp0 + 16);
                a1.q[0] = *(const u32x4*)(hp1);
                a1.q[1] = *(const u32x4*)(hp1 + 16);
                const u32* bp = sWo2p + (ks * 32 + lane) * 8;
                b.q[0] = *(const u32x4*)(bp);
                b.q[1] = *(const u32x4*)(bp + 4);
                c0 = wmma_f16(a0.h, b.h, c0);
                c1 = wmma_f16(a1.h, b.h, c1);
            }
            #pragma unroll
            for (int t = 0; t < 2; t++) {
                #pragma unroll
                for (int v = 0; v < 8; v++) {
                    if (r < O2_F) {
                        float z = t ? c1[v] : c0[v];
                        float s = 1.f / (1.f + __expf(-z));
                        out[(size_t)(tp * 32 + t * 16 + v + kb * 8) * 16 + 10 + r] = s;
                    }
                }
            }
        }
    }
}

extern "C" void kernel_launch(void* const* d_in, const int* in_sizes, int n_in,
                              void* d_out, int out_size, void* d_ws, size_t ws_size,
                              hipStream_t stream) {
    const float* X   = (const float*)d_in[0];
    const float* W1  = (const float*)d_in[1];
    const float* b1  = (const float*)d_in[2];
    const float* Wo1 = (const float*)d_in[3];
    const float* bo1 = (const float*)d_in[4];
    const float* W2  = (const float*)d_in[5];
    const float* b2  = (const float*)d_in[6];
    const float* Wo2 = (const float*)d_in[7];
    const float* bo2 = (const float*)d_in[8];
    float* out = (float*)d_out;

    (void)in_sizes; (void)n_in; (void)out_size; (void)d_ws; (void)ws_size;

    hipFuncSetAttribute((const void*)fused_mlp_kernel,
                        hipFuncAttributeMaxDynamicSharedMemorySize, SMEM_BYTES);
    fused_mlp_kernel<<<NBLOCKS, NTHREADS, SMEM_BYTES, stream>>>(
        X, W1, b1, Wo1, bo1, W2, b2, Wo2, bo2, out);
}
